// LengthRegulator_23880018166299
// MI455X (gfx1250) — matrix operand order repaired and checked by reading, not compiled, not run on hardware
//
#include <hip/hip_runtime.h>

typedef __attribute__((ext_vector_type(16))) __bf16 v16bf;
typedef __attribute__((ext_vector_type(8)))  float  v8f;

#define NB 32
#define LL 1024
#define DD 256
#define TT 4096

// ---------------------------------------------------------------------------
// Swizzle fp32 weights [768][256] into per-lane-contiguous bf16 B-fragments:
// dst[((ntile*24 + kc)*32 + lane)*16 + i] = src[(kc*32 + (i&8)*2 + (lane>>4)*8 + (i&7))*256
//                                               + ntile*16 + (lane&15)]
// (B fragment layout per CDNA5 ISA 7.12.2: K = (i&8)*2 + (lane>>4)*8 + (i&7), N = lane&15)
// ---------------------------------------------------------------------------
__global__ void swizzle_weights_kernel(const float* __restrict__ src,
                                       __bf16* __restrict__ dst) {
    int idx  = blockIdx.x * 256 + threadIdx.x;      // 0 .. 768*256-1
    int i    = idx & 15;
    int lane = (idx >> 4) & 31;
    int kc   = (idx >> 9) % 24;
    int nt   = (idx >> 9) / 24;                     // 0..15
    int K    = kc * 32 + (i & 8) * 2 + (lane >> 4) * 8 + (i & 7);      // 0..767
    int col  = nt * 16 + (lane & 15);               // 0..255
    dst[idx] = (__bf16)src[K * 256 + col];
}

// ---------------------------------------------------------------------------
// Fused conv1d(k=3, SAME) + bias + LayerNorm + ReLU, via bf16 WMMA.
// Block = 256 threads (8 waves); computes 16 L-rows x 256 channels.
// Layer1: writes bf16 h1.  Layer2: additionally dots with Wl -> durations.
// ---------------------------------------------------------------------------
template <bool LAYER2>
__global__ __launch_bounds__(256) void conv_ln_kernel(
    const float*  __restrict__ xf,    // layer1 input (fp32) or null
    const __bf16* __restrict__ xh,    // layer2 input (bf16 h1) or null
    const __bf16* __restrict__ Wt,    // swizzled weights [16][24][32][16]
    const float*  __restrict__ bc,
    const float*  __restrict__ gg,
    const float*  __restrict__ be,
    const float*  __restrict__ Wl,
    const float*  __restrict__ bl,
    __bf16* __restrict__ h1out,
    float*  __restrict__ dur) {
    __shared__ __bf16 xs[18 * 256];     // halo rows l0-1 .. l0+16, bf16
    __shared__ float  Ct[16 * 256];     // conv output tile (f32)
    __shared__ float  mu_s[16], rs_s[16], dsum[16];

    const int n   = blockIdx.y;
    const int l0  = blockIdx.x * 16;
    const int tid = threadIdx.x;

    // Stage input rows (with zero halo) into LDS as bf16.
    for (int r = 0; r < 18; ++r) {
        int l = l0 - 1 + r;
        float v = 0.f;
        if (l >= 0 && l < LL) {
            size_t off = ((size_t)n * LL + l) * 256 + tid;
            v = LAYER2 ? (float)xh[off] : xf[off];
        }
        xs[r * 256 + tid] = (__bf16)v;
    }
    if (tid < 16) dsum[tid] = 0.f;
    __syncthreads();

    const int wave = tid >> 5, lane = tid & 31;
    const int m16 = lane & 15, g = lane >> 4;
    const int nt0 = wave * 2, nt1 = wave * 2 + 1;

    v8f acc0 = {}, acc1 = {};
    for (int kc = 0; kc < 24; ++kc) {            // K = 768 in chunks of 32
        // A fragment: 16-bit A 16x32 layout (ISA 7.12.2)
        v16bf a;
        const int tap = kc >> 3;                 // conv tap 0..2
        const __bf16* xrow = &xs[(m16 + tap) * 256 + (kc & 7) * 32 + g * 8];
#pragma unroll
        for (int j = 0; j < 8; ++j) {
            int klo = (j < 4) ? (j * 2) : (16 + (j - 4) * 2);
            a[2 * j]     = xrow[klo];
            a[2 * j + 1] = xrow[klo + 1];
        }
        // B fragments: contiguous 32B per lane (pre-swizzled)
        v16bf b0 = *(const v16bf*)(Wt + (((size_t)nt0 * 24 + kc) * 32 + lane) * 16);
        v16bf b1 = *(const v16bf*)(Wt + (((size_t)nt1 * 24 + kc) * 32 + lane) * 16);
        acc0 = __builtin_amdgcn_wmma_f32_16x16x32_bf16(false, a, false, b0,
                                                       (short)0, acc0, false, false);
        acc1 = __builtin_amdgcn_wmma_f32_16x16x32_bf16(false, a, false, b1,
                                                       (short)0, acc1, false, false);
    }

    // Scatter C (+ conv bias) into LDS tile. C layout: VGPR r, lanes0-15 M=r,
    // lanes16-31 M=r+8.
    {
        const int cl = lane & 15, row_off = g * 8;
#pragma unroll
        for (int r = 0; r < 8; ++r) {
            int row = r + row_off;
            int c0 = nt0 * 16 + cl, c1 = nt1 * 16 + cl;
            Ct[row * 256 + c0] = acc0[r] + bc[c0];
            Ct[row * 256 + c1] = acc1[r] + bc[c1];
        }
    }
    __syncthreads();

    // LayerNorm stats: 16 threads per row, shfl_xor reduce within 16-lane group.
    {
        const int row = tid >> 4, t16 = tid & 15;
        float s = 0.f, s2 = 0.f;
#pragma unroll
        for (int j = 0; j < 16; ++j) {
            float v = Ct[row * 256 + t16 + j * 16];
            s += v; s2 += v * v;
        }
#pragma unroll
        for (int off = 8; off >= 1; off >>= 1) {
            s  += __shfl_xor(s,  off, 32);
            s2 += __shfl_xor(s2, off, 32);
        }
        if (t16 == 0) {
            float mu  = s * (1.f / 256.f);
            float var = s2 * (1.f / 256.f) - mu * mu;
            mu_s[row] = mu;
            rs_s[row] = rsqrtf(var + 1e-5f);
        }
    }
    __syncthreads();

    const float gc  = gg[tid];
    const float bec = be[tid];
    const float wl  = LAYER2 ? Wl[tid] : 0.f;
#pragma unroll
    for (int e = 0; e < 16; ++e) {
        float v = (Ct[e * 256 + tid] - mu_s[e]) * rs_s[e] * gc + bec;
        v = fmaxf(v, 0.f);
        if (!LAYER2) {
            h1out[((size_t)n * LL + l0 + e) * 256 + tid] = (__bf16)v;
        } else {
            float p = v * wl;
#pragma unroll
            for (int off = 16; off >= 1; off >>= 1) p += __shfl_xor(p, off, 32);
            if (lane == 0) atomicAdd(&dsum[e], p);
        }
    }
    if (LAYER2) {
        __syncthreads();
        if (tid < 16)
            dur[(size_t)n * LL + l0 + tid] = fmaxf(dsum[tid] + bl[0], 0.f);
    }
}

// ---------------------------------------------------------------------------
// Per-batch inclusive cumsum of target[1024] (block scan).
// ---------------------------------------------------------------------------
__global__ __launch_bounds__(256) void cumsum_kernel(const int* __restrict__ target,
                                                     int* __restrict__ cs) {
    __shared__ int wsum[256];
    const int n = blockIdx.x, tid = threadIdx.x;
    int4 v = ((const int4*)(target + (size_t)n * LL))[tid];
    int s0 = v.x, s1 = s0 + v.y, s2 = s1 + v.z, s3 = s2 + v.w;
    wsum[tid] = s3;
    __syncthreads();
    for (int off = 1; off < 256; off <<= 1) {
        int add = (tid >= off) ? wsum[tid - off] : 0;
        __syncthreads();
        wsum[tid] += add;
        __syncthreads();
    }
    int base = wsum[tid] - s3;   // exclusive prefix of this thread's chunk
    int4 o = {base + s0, base + s1, base + s2, base + s3};
    ((int4*)(cs + (size_t)n * LL))[tid] = o;
}

// ---------------------------------------------------------------------------
// Length regulate: one wave per output frame; binary search + float4 row copy.
// ---------------------------------------------------------------------------
__global__ __launch_bounds__(256) void regulate_kernel(const float* __restrict__ x,
                                                       const int* __restrict__ cs,
                                                       float* __restrict__ out) {
    const int n    = blockIdx.y;
    const int wave = threadIdx.x >> 5, lane = threadIdx.x & 31;
    const int t    = blockIdx.x * 8 + wave;
    const int* c   = cs + (size_t)n * LL;
    const int total = c[LL - 1];

    int lo = 0, hi = LL;                 // searchsorted(c, t, side='right')
    while (lo < hi) {
        int mid = (lo + hi) >> 1;
        if (c[mid] <= t) lo = mid + 1; else hi = mid;
    }
    int idx = min(lo, LL - 1);
    bool valid = t < total;

    const float4* src = (const float4*)(x + ((size_t)n * LL + idx) * 256);
    float4* dst = (float4*)(out + ((size_t)n * TT + t) * 256);
    float4 z = {0.f, 0.f, 0.f, 0.f};
    dst[lane * 2]     = valid ? src[lane * 2]     : z;
    dst[lane * 2 + 1] = valid ? src[lane * 2 + 1] : z;
}

// ---------------------------------------------------------------------------
extern "C" void kernel_launch(void* const* d_in, const int* in_sizes, int n_in,
                              void* d_out, int out_size, void* d_ws, size_t ws_size,
                              hipStream_t stream) {
    const float* x      = (const float*)d_in[0];
    const int*   target = (const int*)d_in[1];
    // d_in[2] = mel_max_length (compile-time TT)
    const float* Wc1 = (const float*)d_in[3];
    const float* bc1 = (const float*)d_in[4];
    const float* g1  = (const float*)d_in[5];
    const float* be1 = (const float*)d_in[6];
    const float* Wc2 = (const float*)d_in[7];
    const float* bc2 = (const float*)d_in[8];
    const float* g2  = (const float*)d_in[9];
    const float* be2 = (const float*)d_in[10];
    const float* Wl  = (const float*)d_in[11];
    const float* bl  = (const float*)d_in[12];

    char* ws = (char*)d_ws;
    __bf16* W1t = (__bf16*)(ws);                       // 384 KB
    __bf16* W2t = (__bf16*)(ws + 393216);              // 384 KB
    int*    cs  = (int*)(ws + 786432);                 // 128 KB
    __bf16* h1  = (__bf16*)(ws + 917504);              // 16 MB

    float* out = (float*)d_out;                        // [32][4096][256]
    float* dur = out + (size_t)NB * TT * DD;           // [32][1024]

    swizzle_weights_kernel<<<768, 256, 0, stream>>>(Wc1, W1t);
    swizzle_weights_kernel<<<768, 256, 0, stream>>>(Wc2, W2t);

    dim3 gl(LL / 16, NB);
    conv_ln_kernel<false><<<gl, 256, 0, stream>>>(x, nullptr, W1t, bc1, g1, be1,
                                                  nullptr, nullptr, h1, nullptr);
    conv_ln_kernel<true><<<gl, 256, 0, stream>>>(nullptr, h1, W2t, bc2, g2, be2,
                                                 Wl, bl, nullptr, dur);

    cumsum_kernel<<<NB, 256, 0, stream>>>(target, cs);
    regulate_kernel<<<dim3(TT / 8, NB), 256, 0, stream>>>(x, cs, out);
}
